// GCN_82643760710010
// MI455X (gfx1250) — matrix operand order repaired and checked by reading, not compiled, not run on hardware
//
#include <hip/hip_runtime.h>

// GCN layer on MI455X: out = D A D X W + b, refactored as
//   d = rsqrt(colsum(adj));  S_T[b][o][m] = bf16(d[b,m] * (X W)[b,m,o])
//   out[b,n,o] = d[b,n] * (adj[b,n,:] . S_T[b,o,:]) + bias[o]
// All matrix math via v_wmma_f32_16x16x32_bf16 (fp32 accumulate).

#define NB 8
#define NN 2048
#define NF 256

typedef __attribute__((ext_vector_type(16))) __bf16 v16bf;
typedef __attribute__((ext_vector_type(2)))  __bf16 v2bf;
typedef __attribute__((ext_vector_type(8)))  float  v8f;

union FragU {            // 32 bytes: two b128 loads -> one WMMA operand
  uint4 u4[2];
  v16bf v;
};

// native bf16 converts (v_cvt_pk_bf16_f32 on gfx1250)
__device__ __forceinline__ unsigned cvt2(float a, float b) {
  v2bf v;
  v[0] = (__bf16)a;
  v[1] = (__bf16)b;
  return __builtin_bit_cast(unsigned, v);
}
__device__ __forceinline__ unsigned short cvt1(float a) {
  __bf16 h = (__bf16)a;
  return __builtin_bit_cast(unsigned short, h);
}

// ---------------- kernel 0: weight -> bf16, transposed [o][f] -------------
__global__ void __launch_bounds__(256)
k_prep_wt(const float* __restrict__ w, unsigned short* __restrict__ wt) {
  int t = blockIdx.x * 256 + threadIdx.x;   // 65536 threads
  int o = t & (NF - 1);
  int f = t >> 8;
  wt[o * NF + f] = cvt1(w[f * NF + o]);
}

// ---------------- kernel 1: d[b,j] = rsqrt(sum_i adj[b,i,j]) --------------
__global__ void __launch_bounds__(256)
k_degree(const float* __restrict__ adj, float* __restrict__ d) {
  int t = blockIdx.x * 256 + threadIdx.x;   // NB*NN threads, coalesced in j
  int j = t & (NN - 1);
  int b = t >> 11;
  const float* col = adj + (size_t)b * NN * NN + j;
  float s = 0.f;
#pragma unroll 4
  for (int i = 0; i < NN; ++i) s += col[(size_t)i * NN];
  d[t] = rsqrtf(s);
}

// -------- kernel 2: S_T[b][o][m] = bf16(d[b,m] * (input @ weight)) --------
// 256 thr = 8 waves; block tile 64(m) x 256(o); wave tile 32 x 64 (8 WMMA).
__global__ void __launch_bounds__(256)
k_support(const float* __restrict__ inp, const unsigned short* __restrict__ wt,
          const float* __restrict__ d, unsigned short* __restrict__ st) {
  __shared__ __align__(16) unsigned short lds_a[64 * 48];  // pitch 48 bf16
  const int tid    = threadIdx.x;
  const int lane   = tid & 31;
  const int wave   = tid >> 5;
  const int wave_m = wave >> 2;            // 0..1
  const int wave_n = wave & 3;             // 0..3
  const int b      = blockIdx.x >> 5;      // 32 m-tiles per batch
  const int mBase  = (blockIdx.x & 31) * 64;
  const int lh     = lane >> 4;
  const int l15    = lane & 15;
  const int oCol0  = wave_n * 64;

  v8f acc[2][4] = {};

  for (int f0 = 0; f0 < NF; f0 += 32) {
    __syncthreads();
    {   // stage input tile 64x32 fp32 -> bf16 LDS (coalesced b128 loads)
      int row = tid >> 2;
      int k8  = (tid & 3) * 8;
      const float* src = inp + ((size_t)(b * NN + mBase + row)) * NF + f0 + k8;
      float4 x = *(const float4*)src;
      float4 y = *(const float4*)(src + 4);
      uint4 pk;
      pk.x = cvt2(x.x, x.y);
      pk.y = cvt2(x.z, x.w);
      pk.z = cvt2(y.x, y.y);
      pk.w = cvt2(y.z, y.w);
      *(uint4*)(&lds_a[row * 48 + k8]) = pk;
    }
    __syncthreads();

    FragU a[2];
#pragma unroll
    for (int mt = 0; mt < 2; ++mt) {        // A frag: lanes0-15 K0-15/16-31
      int row = wave_m * 32 + mt * 16 + l15;
      int kc  = lh * 16;
      a[mt].u4[0] = *(const uint4*)(&lds_a[row * 48 + kc]);
      a[mt].u4[1] = *(const uint4*)(&lds_a[row * 48 + kc + 8]);
    }
#pragma unroll
    for (int nt = 0; nt < 4; ++nt) {        // B frag from W_T (contig 32B)
      int o = oCol0 + nt * 16 + l15;
      const unsigned short* bp = wt + o * NF + f0 + lh * 16;
      FragU bf;
      bf.u4[0] = *(const uint4*)bp;
      bf.u4[1] = *(const uint4*)(bp + 8);
#pragma unroll
      for (int mt = 0; mt < 2; ++mt)
        acc[mt][nt] = __builtin_amdgcn_wmma_f32_16x16x32_bf16(
            false, a[mt].v, false, bf.v, (short)0, acc[mt][nt], false, false);
    }
  }

  // epilogue: scale rows by d[b,m]; 8 results per frag are contiguous in
  // S_T[b][o][m0..m0+7] -> pack and emit ONE b128 store per fragment.
#pragma unroll
  for (int mt = 0; mt < 2; ++mt) {
    int m0 = mBase + wave_m * 32 + mt * 16 + lh * 8;   // multiple of 8
    float dv[8];
#pragma unroll
    for (int r = 0; r < 8; ++r) dv[r] = d[b * NN + m0 + r];
#pragma unroll
    for (int nt = 0; nt < 4; ++nt) {
      int o = oCol0 + nt * 16 + l15;
      unsigned short* dst = st + ((size_t)(b * NF + o)) * NN + m0;
      uint4 pk;
      pk.x = cvt2(acc[mt][nt][0] * dv[0], acc[mt][nt][1] * dv[1]);
      pk.y = cvt2(acc[mt][nt][2] * dv[2], acc[mt][nt][3] * dv[3]);
      pk.z = cvt2(acc[mt][nt][4] * dv[4], acc[mt][nt][5] * dv[5]);
      pk.w = cvt2(acc[mt][nt][6] * dv[6], acc[mt][nt][7] * dv[7]);
      *(uint4*)dst = pk;
    }
  }
}

// -------- kernel 3: out[b,n,o] = d[b,n]*(adj[b,n,:] @ S_T[b,o,:]) + bias --
// 512 thr = 16 waves; block tile 64(n) x 256(o); wave tile 32 x 32 (4 WMMA).
// adj streamed from HBM exactly once; LDS double-buffered so the next fp32
// tile load overlaps the WMMAs; one barrier per K-step.
__global__ void __launch_bounds__(512)
k_spmm(const float* __restrict__ adj, const unsigned short* __restrict__ st,
       const float* __restrict__ d, const float* __restrict__ bias,
       float* __restrict__ out) {
  __shared__ __align__(16) unsigned short lds_a[2][64 * 48];
  const int tid    = threadIdx.x;
  const int lane   = tid & 31;
  const int wave   = tid >> 5;
  const int wave_m = wave >> 3;            // 0..1
  const int wave_n = wave & 7;             // 0..7
  const int b      = blockIdx.x >> 5;
  const int nBase  = (blockIdx.x & 31) * 64;
  const int lh     = lane >> 4;
  const int l15    = lane & 15;
  const int oCol0  = wave_n * 32;

  v8f acc[2][2] = {};

  // staging role of this thread: one float4 (4 K-values) of one adj row
  const int srow = tid >> 3;
  const int sk4  = (tid & 7) * 4;
  const float* src0 =
      adj + (size_t)b * NN * NN + (size_t)(nBase + srow) * NN + sk4;

  // prologue: stage tile for k0 = 0 into buffer 0
  {
    float4 x = *(const float4*)src0;
    uint2 pk;
    pk.x = cvt2(x.x, x.y);
    pk.y = cvt2(x.z, x.w);
    *(uint2*)(&lds_a[0][srow * 48 + sk4]) = pk;
  }
  __syncthreads();

  int buf = 0;
  for (int k0 = 0; k0 < NN; k0 += 32) {
    const bool have_next = (k0 + 32 < NN);
    float4 nx;
    if (have_next) {                         // HBM load overlaps WMMAs below
      nx = *(const float4*)(src0 + k0 + 32);
      if (k0 + 64 < NN) __builtin_prefetch(src0 + k0 + 64, 0, 1);
    }

    FragU a[2];
#pragma unroll
    for (int mt = 0; mt < 2; ++mt) {
      int row = wave_m * 32 + mt * 16 + l15;
      int kc  = lh * 16;
      a[mt].u4[0] = *(const uint4*)(&lds_a[buf][row * 48 + kc]);
      a[mt].u4[1] = *(const uint4*)(&lds_a[buf][row * 48 + kc + 8]);
    }
#pragma unroll
    for (int nt = 0; nt < 2; ++nt) {        // B from L2-resident S_T
      int o = oCol0 + nt * 16 + l15;
      const unsigned short* bp =
          st + ((size_t)(b * NF + o)) * NN + k0 + lh * 16;
      FragU bf;
      bf.u4[0] = *(const uint4*)bp;
      bf.u4[1] = *(const uint4*)(bp + 8);
#pragma unroll
      for (int mt = 0; mt < 2; ++mt)
        acc[mt][nt] = __builtin_amdgcn_wmma_f32_16x16x32_bf16(
            false, a[mt].v, false, bf.v, (short)0, acc[mt][nt], false, false);
    }

    if (have_next) {                         // write next tile to other buffer
      uint2 pk;
      pk.x = cvt2(nx.x, nx.y);
      pk.y = cvt2(nx.z, nx.w);
      *(uint2*)(&lds_a[buf ^ 1][srow * 48 + sk4]) = pk;
    }
    __syncthreads();                         // single barrier per K-step
    buf ^= 1;
  }

  // epilogue: out = d[b,n]*acc + bias[o]  (fp32 store)
#pragma unroll
  for (int mt = 0; mt < 2; ++mt) {
    int n0 = nBase + wave_m * 32 + mt * 16 + lh * 8;
    float dv[8];
#pragma unroll
    for (int r = 0; r < 8; ++r) dv[r] = d[b * NN + n0 + r];
#pragma unroll
    for (int nt = 0; nt < 2; ++nt) {
      int o  = oCol0 + nt * 16 + l15;
      float bo = bias[o];
      float* dst = out + ((size_t)(b * NN + n0)) * NF + o;
#pragma unroll
      for (int r = 0; r < 8; ++r)
        dst[(size_t)r * NF] = acc[mt][nt][r] * dv[r] + bo;
    }
  }
}

extern "C" void kernel_launch(void* const* d_in, const int* in_sizes, int n_in,
                              void* d_out, int out_size, void* d_ws, size_t ws_size,
                              hipStream_t stream) {
  const float* inp  = (const float*)d_in[0];   // [8,2048,256]
  const float* adj  = (const float*)d_in[1];   // [8,2048,2048]
  const float* w    = (const float*)d_in[2];   // [256,256]
  const float* bias = (const float*)d_in[3];   // [256]
  float* out = (float*)d_out;                  // [8,2048,256] fp32

  char* ws = (char*)d_ws;
  float*          d_deg = (float*)ws;                            // 64 KB
  unsigned short* wt    = (unsigned short*)(ws + (64u << 10));   // 128 KB
  unsigned short* st    = (unsigned short*)(ws + (192u << 10));  // 8 MB

  k_prep_wt<<<NF * NF / 256, 256, 0, stream>>>(w, wt);
  k_degree <<<NB * NN / 256, 256, 0, stream>>>(adj, d_deg);
  k_support<<<NB * (NN / 64), 256, 0, stream>>>(inp, wt, d_deg, st);
  k_spmm   <<<NB * (NN / 64), 512, 0, stream>>>(adj, st, d_deg, bias, out);
}
